// QuantumMotivicTile_33603824124390
// MI455X (gfx1250) — compile-verified, matrix operand order checked by reading, not run on hardware
//
#include <hip/hip_runtime.h>
#include <hip/hip_bf16.h>

// ---------------------------------------------------------------------------
// MI455X (gfx1250, wave32) implementation.
// Reference = 16-head attention (S=2048, hd=64) with linear pre/post stages
// that algebraically fold into single 64x64 effective weights.
// All GEMMs run on v_wmma_f32_16x16x32_f16 (f16 in, f32 accumulate).
// K/V chunks are double-buffered in LDS via GLOBAL_LOAD_ASYNC_TO_LDS_B128
// when the toolchain exposes it (guarded; synchronous fallback otherwise).
// ---------------------------------------------------------------------------

typedef _Float16 v16h __attribute__((ext_vector_type(16)));
typedef _Float16 v8h  __attribute__((ext_vector_type(8)));
typedef float    v8f  __attribute__((ext_vector_type(8)));
typedef int      v4i  __attribute__((ext_vector_type(4)));

#define B_DIM 4
#define S_LEN 2048
#define NH    16
#define HD    64
#define LDP   80   // padded LDS row stride in halves (160B, 16B-aligned, bank-skewed)

#define GLOBAL_AS __attribute__((address_space(1)))
#define LDS_AS    __attribute__((address_space(3)))

#if __has_builtin(__builtin_amdgcn_global_load_async_to_lds_b128)
#define HAVE_ASYNC_LDS 1
#else
#define HAVE_ASYNC_LDS 0
#endif

__device__ __forceinline__ v8f wmma_acc(v16h a, v16h b, v8f c) {
  // (neg_a, A, neg_b, B, c_mod, C, reuse_a, reuse_b)
  return __builtin_amdgcn_wmma_f32_16x16x32_f16(false, a, false, b, (short)0, c,
                                                false, false);
}

// 16-byte global->LDS copy: async DMA path when available, else vector ld/st.
__device__ __forceinline__ void cp16_g2l(const _Float16* g, _Float16* l) {
#if HAVE_ASYNC_LDS
  __builtin_amdgcn_global_load_async_to_lds_b128((GLOBAL_AS v4i*)g,
                                                 (LDS_AS v4i*)l, 0, 0);
#else
  *(v8h*)l = *(const v8h*)g;
#endif
}

__device__ __forceinline__ void wait_async_all() {
#if HAVE_ASYNC_LDS
#if __has_builtin(__builtin_amdgcn_s_wait_asynccnt)
  __builtin_amdgcn_s_wait_asynccnt(0);
#else
  asm volatile("s_wait_asynccnt 0x0" ::: "memory");
#endif
#endif
}

// A/B fragment gather from a row-major f16 matrix (ISA 7.12.2 16-bit 16x32):
// lane&15 selects the row (M for A, N for B when source is [N][K] row-major);
// halves 0..7 -> K = k0+{0..7}, halves 8..15 -> K = k0+{16..23}; lanes>=16 add 8.
__device__ __forceinline__ v16h load_frag16(const _Float16* base, int ld, int r0,
                                            int k0, int lane) {
  const _Float16* p =
      base + (size_t)(r0 + (lane & 15)) * ld + k0 + ((lane & 16) ? 8 : 0);
  v8h lo = *(const v8h*)p;
  v8h hi = *(const v8h*)(p + 16);
  v16h r;
#pragma unroll
  for (int i = 0; i < 8; ++i) { r[i] = lo[i]; r[i + 8] = hi[i]; }
  return r;
}

// Same gather from f32 source with convert-to-f16.
__device__ __forceinline__ v16h load_frag32(const float* base, int ld, int r0,
                                            int k0, int lane) {
  const float* p =
      base + (size_t)(r0 + (lane & 15)) * ld + k0 + ((lane & 16) ? 8 : 0);
  v16h r;
#pragma unroll
  for (int i = 0; i < 8; ++i) {
    r[i]     = (_Float16)p[i];
    r[i + 8] = (_Float16)p[i + 16];
  }
  return r;
}

// ---------------------------------------------------------------------------
// Kernel 1: fold Wi into Wq/Wk/Wv (and 1/sqrt(hd) into the Q pair); collapse
// cohomology+motive round trips into one 64x64 Weff and bias beff.
// One block, 64 threads (thread = output row).
// ---------------------------------------------------------------------------
__global__ void __launch_bounds__(64) prep_weights_kernel(
    const float* __restrict__ Wi,  const float* __restrict__ bi,
    const float* __restrict__ Wq,  const float* __restrict__ bq,
    const float* __restrict__ Wk,  const float* __restrict__ bk,
    const float* __restrict__ Wv,  const float* __restrict__ bv,
    const float* __restrict__ Wc,  const float* __restrict__ bc,
    const float* __restrict__ Wci, const float* __restrict__ bci,
    const float* __restrict__ Wm,  const float* __restrict__ bm,
    const float* __restrict__ Wmi, const float* __restrict__ bmi,
    _Float16* __restrict__ WqE, _Float16* __restrict__ WkE,
    _Float16* __restrict__ WvE, _Float16* __restrict__ WeE,
    float* __restrict__ bqE, float* __restrict__ bkE,
    float* __restrict__ bvE, float* __restrict__ beE) {
  const int o = threadIdx.x;  // 0..63
  const float qscale = 0.125f;  // 1/sqrt(64), folded into Q projection

  // W' = Wx @ Wi  (q = x @ (Wq Wi)^T + (Wq bi + bq))
  for (int i = 0; i < 64; ++i) {
    float sq = 0.f, sk = 0.f, sv = 0.f;
    for (int j = 0; j < 64; ++j) {
      const float wij = Wi[j * 64 + i];
      sq += Wq[o * 64 + j] * wij;
      sk += Wk[o * 64 + j] * wij;
      sv += Wv[o * 64 + j] * wij;
    }
    WqE[o * 64 + i] = (_Float16)(sq * qscale);
    WkE[o * 64 + i] = (_Float16)sk;
    WvE[o * 64 + i] = (_Float16)sv;
  }
  {
    float tq = bq[o], tk = bk[o], tv = bv[o];
    for (int j = 0; j < 64; ++j) {
      const float bij = bi[j];
      tq += Wq[o * 64 + j] * bij;
      tk += Wk[o * 64 + j] * bij;
      tv += Wv[o * 64 + j] * bij;
    }
    bqE[o] = tq * qscale; bkE[o] = tk; bvE[o] = tv;
  }

  // T1 = Wci @ Wc (rank 8), T2 = Wmi @ Wm (rank 4), Weff = T2 @ T1.
  __shared__ float T1[64 * 64];
  __shared__ float T2[64 * 64];
  __shared__ float b1s[64];
  for (int j = 0; j < 64; ++j) {
    float s1 = 0.f;
    for (int c = 0; c < 8; ++c) s1 += Wci[o * 8 + c] * Wc[c * 64 + j];
    T1[o * 64 + j] = s1;
    float s2 = 0.f;
    for (int m = 0; m < 4; ++m) s2 += Wmi[o * 4 + m] * Wm[m * 64 + j];
    T2[o * 64 + j] = s2;
  }
  {
    float s = bci[o];
    for (int c = 0; c < 8; ++c) s += Wci[o * 8 + c] * bc[c];
    b1s[o] = s;
  }
  __syncthreads();
  for (int j = 0; j < 64; ++j) {
    float s = 0.f;
    for (int k = 0; k < 64; ++k) s += T2[o * 64 + k] * T1[k * 64 + j];
    WeE[o * 64 + j] = (_Float16)s;
  }
  // beff = bmi + Wmi @ (bm + Wm @ b1)
  float be_o = bmi[o];
  for (int m = 0; m < 4; ++m) {
    float b2 = bm[m];
    for (int j = 0; j < 64; ++j) b2 += Wm[m * 64 + j] * b1s[j];
    be_o += Wmi[o * 4 + m] * b2;
  }
  beE[o] = be_o;
}

// One 16x64 output slice: C = A(16x64) x W^T via 4 N-tiles x 2 K-steps.
// VT=false: store row-major [s][d]. VT=true: store transposed [d][s]
// (used for V so attention staging is a contiguous copy for both K and V).
template <bool VT>
__device__ __forceinline__ void project_store(v16h a0, v16h a1,
                                              const _Float16* __restrict__ W,
                                              const float* __restrict__ bias,
                                              _Float16* __restrict__ dst,
                                              int srow, int lane) {
  const int prow = (lane & 16) ? 8 : 0;
#pragma unroll
  for (int n = 0; n < 4; ++n) {
    v16h b0 = load_frag16(W, HD, n * 16, 0, lane);
    v16h b1 = load_frag16(W, HD, n * 16, 32, lane);
    v8f c = {};
    c = wmma_acc(a0, b0, c);
    c = wmma_acc(a1, b1, c);
    const int col = n * 16 + (lane & 15);
    const float bb = bias[col];
#pragma unroll
    for (int r = 0; r < 8; ++r) {
      const int row = srow + r + prow;
      const size_t idx = VT ? (size_t)col * S_LEN + row : (size_t)row * HD + col;
      dst[idx] = (_Float16)(c[r] + bb);
    }
  }
}

// ---------------------------------------------------------------------------
// Kernel 2: Q/K/V projection (input fold pre-applied). grid = B*H*(S/64),
// block = 128 (4 waves), each wave owns a 16-row M-slice. V stored [d][s].
// ---------------------------------------------------------------------------
__global__ void __launch_bounds__(128) qkv_project_kernel(
    const float* __restrict__ x,
    const _Float16* __restrict__ WqE, const _Float16* __restrict__ WkE,
    const _Float16* __restrict__ WvE,
    const float* __restrict__ bqE, const float* __restrict__ bkE,
    const float* __restrict__ bvE,
    _Float16* __restrict__ Qg, _Float16* __restrict__ Kg,
    _Float16* __restrict__ Vtg) {
  const int lane = threadIdx.x & 31;
  const int wave = threadIdx.x >> 5;
  const int tile = blockIdx.x;
  const int st = tile & 31;        // 32 tiles of 64 rows over S
  const int bh = tile >> 5;        // b*NH + h
  const int h = bh & (NH - 1);
  const int b = bh >> 4;
  const int srow = st * 64 + wave * 16;

  const float* xh = x + (size_t)b * S_LEN * 1024 + (size_t)h * HD;
  const v16h a0 = load_frag32(xh, 1024, srow, 0, lane);
  const v16h a1 = load_frag32(xh, 1024, srow, 32, lane);

  const size_t headoff = (size_t)bh * S_LEN * HD;
  project_store<false>(a0, a1, WqE, bqE, Qg + headoff, srow, lane);
  project_store<false>(a0, a1, WkE, bkE, Kg + headoff, srow, lane);
  project_store<true >(a0, a1, WvE, bvE, Vtg + headoff, srow, lane);
}

// ---------------------------------------------------------------------------
// Kernel 3: flash attention + fused collapsed post-projection + head merge.
// grid = B*H*(S/64), block = 128 (4 waves). Wave owns a 16-row q slice;
// K (row-major) and V^T ([d][s]) streamed in 64-row chunks through
// double-buffered LDS, staged with async global->LDS copies when available.
// ---------------------------------------------------------------------------
__global__ void __launch_bounds__(128) motivic_attn_kernel(
    const _Float16* __restrict__ Qg, const _Float16* __restrict__ Kg,
    const _Float16* __restrict__ Vtg,
    const _Float16* __restrict__ WeE, const float* __restrict__ beE,
    float* __restrict__ out) {
  __shared__ __align__(16) _Float16 sK[2][64 * LDP];   // K chunk, row-major
  __shared__ __align__(16) _Float16 sVt[2][64 * LDP];  // V^T chunk: [d][j]
  __shared__ __align__(16) _Float16 sP[4][16 * LDP];   // per-wave P/O staging

  const int lane = threadIdx.x & 31;
  const int wave = threadIdx.x >> 5;
  const int tile = blockIdx.x;
  const int qt = tile & 31;
  const int bh = tile >> 5;
  const int h = bh & (NH - 1);
  const int b = bh >> 4;

  const size_t headoff = (size_t)bh * S_LEN * HD;
  const _Float16* Qh = Qg + headoff;
  const _Float16* Kh = Kg + headoff;
  const _Float16* Vth = Vtg + headoff;

  const int qrow = qt * 64 + wave * 16;
  const int prow = (lane & 16) ? 8 : 0;

  // Q fragments (1/sqrt(hd) already folded into the projection).
  const v16h aq0 = load_frag16(Qh, HD, qrow, 0, lane);
  const v16h aq1 = load_frag16(Qh, HD, qrow, 32, lane);

  v8f oacc[4] = {};
  float m_r[8], l_r[8];
#pragma unroll
  for (int r = 0; r < 8; ++r) { m_r[r] = -3.0e38f; l_r[r] = 0.0f; }

  _Float16* pw = sP[wave];

  // Stage one 64-row chunk of K and V^T into LDS buffer `buf`.
  auto stage = [&](int buf, int kc) {
    for (int i = threadIdx.x; i < 64 * 8; i += 128) {
      const int row = i >> 3;
      const int c8 = (i & 7) * 8;
      cp16_g2l(&Kh[(size_t)(kc + row) * HD + c8], &sK[buf][row * LDP + c8]);
      cp16_g2l(&Vth[(size_t)row * S_LEN + kc + c8], &sVt[buf][row * LDP + c8]);
    }
  };

  stage(0, 0);
  wait_async_all();
  __syncthreads();

  for (int kc = 0; kc < S_LEN; kc += 64) {
    const int cur = (kc >> 6) & 1;
    const bool has_next = (kc + 64) < S_LEN;
    if (has_next) stage(cur ^ 1, kc + 64);  // overlap DMA with compute

    // S-tile = (Q/sqrt(hd)) K^T ; 4 N-tiles of 16 k-rows each.
    v8f sf[4];
#pragma unroll
    for (int n = 0; n < 4; ++n) {
      v16h bk0 = load_frag16(sK[cur], LDP, n * 16, 0, lane);
      v16h bk1 = load_frag16(sK[cur], LDP, n * 16, 32, lane);
      v8f c = {};
      c = wmma_acc(aq0, bk0, c);
      c = wmma_acc(aq1, bk1, c);
      sf[n] = c;
    }

    // Online softmax. C-frag row r lives on the 16 lanes of this lane's half,
    // so row reductions are shfl_xor over masks 1,2,4,8 (stay within half).
#pragma unroll
    for (int r = 0; r < 8; ++r) {
      float mx = fmaxf(fmaxf(sf[0][r], sf[1][r]), fmaxf(sf[2][r], sf[3][r]));
#pragma unroll
      for (int off = 1; off < 16; off <<= 1) mx = fmaxf(mx, __shfl_xor(mx, off, 32));
      const float mnew = fmaxf(m_r[r], mx);
      const float corr = __expf(m_r[r] - mnew);
      float ps = 0.0f;
#pragma unroll
      for (int n = 0; n < 4; ++n) {
        const float p = __expf(sf[n][r] - mnew);
        sf[n][r] = p;
        ps += p;
      }
#pragma unroll
      for (int off = 1; off < 16; off <<= 1) ps += __shfl_xor(ps, off, 32);
      l_r[r] = l_r[r] * corr + ps;
      m_r[r] = mnew;
#pragma unroll
      for (int n = 0; n < 4; ++n) oacc[n][r] *= corr;
    }

    // Stage P (f16) wave-locally, reload in A-fragment layout.
#pragma unroll
    for (int n = 0; n < 4; ++n)
#pragma unroll
      for (int r = 0; r < 8; ++r)
        pw[(r + prow) * LDP + n * 16 + (lane & 15)] = (_Float16)sf[n][r];

    v16h ap0 = load_frag16(pw, LDP, 0, 0, lane);
    v16h ap1 = load_frag16(pw, LDP, 0, 32, lane);

    // O += P @ V  (B-fragments straight out of the V^T chunk).
#pragma unroll
    for (int n = 0; n < 4; ++n) {
      v16h bv0 = load_frag16(sVt[cur], LDP, n * 16, 0, lane);
      v16h bv1 = load_frag16(sVt[cur], LDP, n * 16, 32, lane);
      oacc[n] = wmma_acc(ap0, bv0, oacc[n]);
      oacc[n] = wmma_acc(ap1, bv1, oacc[n]);
    }

    if (has_next) wait_async_all();  // own async copies done...
    __syncthreads();                 // ...and everyone's; cur fully consumed
  }

  // Normalize.
#pragma unroll
  for (int r = 0; r < 8; ++r) {
    const float inv = 1.0f / l_r[r];
#pragma unroll
    for (int n = 0; n < 4; ++n) oacc[n][r] *= inv;
  }

  // Fused epilogue: y = O @ Weff^T + beff, merged-head store to [B,S,1024].
#pragma unroll
  for (int n = 0; n < 4; ++n)
#pragma unroll
    for (int r = 0; r < 8; ++r)
      pw[(r + prow) * LDP + n * 16 + (lane & 15)] = (_Float16)oacc[n][r];

  v16h ao0 = load_frag16(pw, LDP, 0, 0, lane);
  v16h ao1 = load_frag16(pw, LDP, 0, 32, lane);

  float* outh = out + (size_t)b * S_LEN * 1024 + (size_t)h * HD;
#pragma unroll
  for (int n = 0; n < 4; ++n) {
    v16h bw0 = load_frag16(WeE, HD, n * 16, 0, lane);
    v16h bw1 = load_frag16(WeE, HD, n * 16, 32, lane);
    v8f y = {};
    y = wmma_acc(ao0, bw0, y);
    y = wmma_acc(ao1, bw1, y);
    const int col = n * 16 + (lane & 15);
    const float bias = beE[col];
#pragma unroll
    for (int r = 0; r < 8; ++r)
      outh[(size_t)(qrow + r + prow) * 1024 + col] = y[r] + bias;
  }
}

// ---------------------------------------------------------------------------
extern "C" void kernel_launch(void* const* d_in, const int* in_sizes, int n_in,
                              void* d_out, int out_size, void* d_ws, size_t ws_size,
                              hipStream_t stream) {
  const float* x   = (const float*)d_in[0];
  const float* Wi  = (const float*)d_in[1];
  const float* bi  = (const float*)d_in[2];
  const float* Wq  = (const float*)d_in[3];
  const float* bq  = (const float*)d_in[4];
  const float* Wk  = (const float*)d_in[5];
  const float* bk  = (const float*)d_in[6];
  const float* Wv  = (const float*)d_in[7];
  const float* bv  = (const float*)d_in[8];
  const float* Wc  = (const float*)d_in[9];
  const float* bc  = (const float*)d_in[10];
  const float* Wci = (const float*)d_in[11];
  const float* bci = (const float*)d_in[12];
  const float* Wm  = (const float*)d_in[13];
  const float* bm  = (const float*)d_in[14];
  const float* Wmi = (const float*)d_in[15];
  const float* bmi = (const float*)d_in[16];

  // Workspace carving (~48.1 MB): effective f16 weights + f32 biases +
  // f16 Q/K staging [B*H, S, HD] and V^T staging [B*H, HD, S].
  char* w = (char*)d_ws;
  auto take = [&](size_t bytes) {
    char* p = w;
    w += (bytes + 255) & ~(size_t)255;
    return p;
  };
  _Float16* WqE = (_Float16*)take(64 * 64 * sizeof(_Float16));
  _Float16* WkE = (_Float16*)take(64 * 64 * sizeof(_Float16));
  _Float16* WvE = (_Float16*)take(64 * 64 * sizeof(_Float16));
  _Float16* WeE = (_Float16*)take(64 * 64 * sizeof(_Float16));
  float* bqE = (float*)take(64 * sizeof(float));
  float* bkE = (float*)take(64 * sizeof(float));
  float* bvE = (float*)take(64 * sizeof(float));
  float* beE = (float*)take(64 * sizeof(float));
  const size_t qkv_bytes = (size_t)B_DIM * NH * S_LEN * HD * sizeof(_Float16);
  _Float16* Qg  = (_Float16*)take(qkv_bytes);
  _Float16* Kg  = (_Float16*)take(qkv_bytes);
  _Float16* Vtg = (_Float16*)take(qkv_bytes);

  prep_weights_kernel<<<1, 64, 0, stream>>>(
      Wi, bi, Wq, bq, Wk, bk, Wv, bv, Wc, bc, Wci, bci, Wm, bm, Wmi, bmi,
      WqE, WkE, WvE, WeE, bqE, bkE, bvE, beE);

  const int tiles = B_DIM * NH * (S_LEN / 64);  // 2048
  qkv_project_kernel<<<tiles, 128, 0, stream>>>(
      x, WqE, WkE, WvE, bqE, bkE, bvE, Qg, Kg, Vtg);

  motivic_attn_kernel<<<tiles, 128, 0, stream>>>(
      Qg, Kg, Vtg, WeE, beE, (float*)d_out);
}
